// Head_62285615727159
// MI455X (gfx1250) — 8-bit matrix operand order repaired and checked by reading, compiled, not run on hardware
//
#include <hip/hip_runtime.h>


typedef __attribute__((ext_vector_type(16))) _Float16     v16h;
typedef __attribute__((ext_vector_type(8)))  float        v8f;
typedef __attribute__((ext_vector_type(8)))  int          v8i;
typedef __attribute__((ext_vector_type(4)))  int          v4i;
typedef __attribute__((ext_vector_type(4)))  unsigned int v4u;

static constexpr int    Bb  = 8;
static constexpr int    Tt  = 2048;
static constexpr int    Cc  = 1024;
static constexpr int    Hh  = 64;
static constexpr int    BT  = Bb * Tt;
static constexpr float  EPS = 1e-5f;
static constexpr int    WBYTES = 3 * Hh * Cc;

static constexpr size_t OFF_W8    = 0;
static constexpr size_t OFF_WMEAN = 196608;
static constexpr size_t OFF_AMAX  = 196864;
static constexpr size_t OFF_XQ    = 262400;
static constexpr size_t OFF_QF    = 17039616;
static constexpr size_t OFF_KF    = 19136768;
static constexpr size_t OFF_VT    = 21233920;

__global__ __launch_bounds__(256) void wquant_kernel(
    const float* __restrict__ wk, const float* __restrict__ wq,
    const float* __restrict__ wv, signed char* __restrict__ w8,
    float* __restrict__ wmean) {
  const int mat = blockIdx.x;
  const float* w = (mat == 0) ? wk : (mat == 1) ? wq : wv;
  __shared__ float red[256];
  float s = 0.f;
  for (int i = threadIdx.x; i < Hh * Cc; i += 256) s += fabsf(w[i]);
  red[threadIdx.x] = s;
  __syncthreads();
  for (int off = 128; off > 0; off >>= 1) {
    if ((int)threadIdx.x < off) red[threadIdx.x] += red[threadIdx.x + off];
    __syncthreads();
  }
  const float mean  = red[0] * (1.0f / (Hh * Cc));
  const float m     = fmaxf(mean, EPS);
  const float scale = 1.0f / m;
  if (threadIdx.x == 0) wmean[mat] = m;
  signed char* dst = w8 + (size_t)mat * Hh * Cc;
  for (int i = threadIdx.x; i < Hh * Cc; i += 256) {
    int t = (int)rintf(w[i] * scale);
    t = (t < -1) ? -1 : (t > 1) ? 1 : t;
    dst[i] = (signed char)t;
  }
}

__global__ __launch_bounds__(256) void actquant_kernel(
    const float* __restrict__ x, signed char* __restrict__ xq,
    float* __restrict__ amax) {
  const int tok = blockIdx.x;
  const int tid = threadIdx.x;
  const float4 v = ((const float4*)(x + (size_t)tok * Cc))[tid];
  float a = fmaxf(fmaxf(fabsf(v.x), fabsf(v.y)), fmaxf(fabsf(v.z), fabsf(v.w)));
  __shared__ float red[256];
  red[tid] = a;
  __syncthreads();
  for (int off = 128; off > 0; off >>= 1) {
    if (tid < off) red[tid] = fmaxf(red[tid], red[tid + off]);
    __syncthreads();
  }
  const float am    = fmaxf(red[0], EPS);
  const float scale = 127.0f / am;
  if (tid == 0) amax[tok] = am;
  int q0 = (int)rintf(v.x * scale); q0 = q0 < -128 ? -128 : (q0 > 127 ? 127 : q0);
  int q1 = (int)rintf(v.y * scale); q1 = q1 < -128 ? -128 : (q1 > 127 ? 127 : q1);
  int q2 = (int)rintf(v.z * scale); q2 = q2 < -128 ? -128 : (q2 > 127 ? 127 : q2);
  int q3 = (int)rintf(v.w * scale); q3 = q3 < -128 ? -128 : (q3 > 127 ? 127 : q3);
  const int word = (q0 & 255) | ((q1 & 255) << 8) | ((q2 & 255) << 16) | ((q3 & 255) << 24);
  ((int*)(xq + (size_t)tok * Cc))[tid] = word;
}

__global__ __launch_bounds__(256) void qkv_kernel(
    const signed char* __restrict__ xq, const float* __restrict__ amax,
    const signed char* __restrict__ w8, const float* __restrict__ wmean,
    _Float16* __restrict__ Kf, _Float16* __restrict__ Qf,
    _Float16* __restrict__ Vt) {
  extern __shared__ char wlds_raw[];
  signed char* wlds = (signed char*)wlds_raw;

  const int wave = threadIdx.x >> 5;
  const int lane = threadIdx.x & 31;
  const int n16  = lane & 15;
  const int g    = lane >> 4;
  const int tokenBase = (blockIdx.x * 8 + wave) * 16;

#if __has_builtin(__builtin_amdgcn_tensor_load_to_lds)
  if (wave == 0) {
    const unsigned long long ga = (unsigned long long)(uintptr_t)w8;
    const unsigned nelem = (unsigned)(WBYTES >> 3);
    v4u g0;
    g0[0] = 1u;
    g0[1] = 0u;
    g0[2] = (unsigned)(ga & 0xFFFFFFFFu);
    g0[3] = (unsigned)((ga >> 32) & 0x01FFFFFFu) | 0x80000000u;
    v8i g1 = {};
    g1[0] = (int)(3u << 16);
    g1[1] = (int)((nelem & 0xFFFFu) << 16);
    g1[2] = (int)(((nelem >> 16) & 0xFFFFu) | (1u << 16));
    g1[3] = (int)((nelem & 0xFFFFu) << 16);
    g1[5] = (int)nelem;
    v4i g2 = {};
    v4i g3 = {};
#if defined(__clang_major__) && (__clang_major__ >= 23)
    v8i g4 = {};
    __builtin_amdgcn_tensor_load_to_lds(g0, g1, g2, g3, g4, 0);
#else
    __builtin_amdgcn_tensor_load_to_lds(g0, g1, g2, g3, 0);
#endif
    __builtin_amdgcn_s_wait_tensorcnt(0);
  }
#else
  for (int i = threadIdx.x * 16; i < WBYTES; i += 256 * 16)
    *(int4*)(wlds + i) = *(const int4*)(w8 + i);
#endif
  __syncthreads();

  const signed char* xrow = xq + (size_t)(tokenBase + n16) * Cc;

  v8i acc[12] = {};
#pragma unroll
  for (int k0 = 0; k0 < Cc; k0 += 64) {
    __builtin_prefetch(xrow + k0 + 128, 0, 1);
    v8i af;
#pragma unroll
    for (int c = 0; c < 4; c++) {
      const uint2 u = *(const uint2*)(xrow + k0 + 8 * c + 16 * g + (c >= 2 ? 16 : 0));
      af[2 * c]     = (int)u.x;
      af[2 * c + 1] = (int)u.y;
    }
#pragma unroll
    for (int mat = 0; mat < 3; mat++) {
#pragma unroll
      for (int ct = 0; ct < 4; ct++) {
        const signed char* wrow = wlds + ((size_t)mat * Hh + 16 * ct + n16) * Cc;
        const int4 b0 = *(const int4*)(wrow + k0 + 16 * g);
        const int4 b1 = *(const int4*)(wrow + k0 + 32 + 16 * g);
        v8i bf;
        bf[0] = b0.x; bf[1] = b0.y; bf[2] = b0.z; bf[3] = b0.w;
        bf[4] = b1.x; bf[5] = b1.y; bf[6] = b1.z; bf[7] = b1.w;
        acc[mat * 4 + ct] = __builtin_amdgcn_wmma_i32_16x16x64_iu8(
            true, af, true, bf, acc[mat * 4 + ct], false, false);
      }
    }
  }

  float ar[8];
  const float4 a0 = *(const float4*)(amax + tokenBase + 8 * g);
  const float4 a1 = *(const float4*)(amax + tokenBase + 8 * g + 4);
  ar[0] = a0.x; ar[1] = a0.y; ar[2] = a0.z; ar[3] = a0.w;
  ar[4] = a1.x; ar[5] = a1.y; ar[6] = a1.z; ar[7] = a1.w;
#pragma unroll
  for (int i = 0; i < 8; i++) ar[i] *= (1.0f / 127.0f);
  const float wm[3] = {wmean[0], wmean[1], wmean[2]};

#pragma unroll
  for (int mat = 0; mat < 3; mat++) {
#pragma unroll
    for (int ct = 0; ct < 4; ct++) {
#pragma unroll
      for (int i = 0; i < 8; i++) {
        const int token = tokenBase + 8 * g + i;
        const int col   = 16 * ct + n16;
        const float f   = (float)acc[mat * 4 + ct][i] * ar[i] * wm[mat];
        const _Float16 h = (_Float16)f;
        if (mat == 0)      Kf[(size_t)token * Hh + col] = h;
        else if (mat == 1) Qf[(size_t)token * Hh + col] = h;
        else {
          const int bb = token >> 11;
          const int tt = token & (Tt - 1);
          Vt[((size_t)bb * Hh + col) * Tt + tt] = h;
        }
      }
    }
  }
}

__global__ __launch_bounds__(256) void attn_kernel(
    const _Float16* __restrict__ Qf, const _Float16* __restrict__ Kf,
    const _Float16* __restrict__ Vt, float* __restrict__ out) {
  __shared__ __align__(16) _Float16 plds[8][16][32];

  const int wave = threadIdx.x >> 5;
  const int lane = threadIdx.x & 31;
  const int n16  = lane & 15;
  const int g    = lane >> 4;
  const int b    = blockIdx.x >> 4;
  const int qt   = blockIdx.x & 15;
  const int qBase = qt * 128 + wave * 16;

  const _Float16* qrow = Qf + ((size_t)b * Tt + qBase + n16) * Hh;
  v16h qa0, qa1;
  *(float4*)&qa0       = *(const float4*)(qrow + 8 * g);
  *((float4*)&qa0 + 1) = *(const float4*)(qrow + 16 + 8 * g);
  *(float4*)&qa1       = *(const float4*)(qrow + 32 + 8 * g);
  *((float4*)&qa1 + 1) = *(const float4*)(qrow + 48 + 8 * g);

  v8f   o[4] = {};
  float mrow[8], lsum[8];
#pragma unroll
  for (int i = 0; i < 8; i++) { mrow[i] = -1e30f; lsum[i] = 0.f; }

  const int nkb = qBase / 32 + 1;
  for (int jb = 0; jb < nkb; jb++) {
    const int  jBase = jb * 32;
    const bool lastb = (jb == nkb - 1);

    __builtin_prefetch(Kf + ((size_t)b * Tt + jBase + 32 + n16) * Hh, 0, 1);

    v16h kb[4];
#pragma unroll
    for (int sub = 0; sub < 2; sub++) {
      const _Float16* krow = Kf + ((size_t)b * Tt + jBase + sub * 16 + n16) * Hh;
      *(float4*)&kb[sub * 2]           = *(const float4*)(krow + 8 * g);
      *((float4*)&kb[sub * 2] + 1)     = *(const float4*)(krow + 16 + 8 * g);
      *(float4*)&kb[sub * 2 + 1]       = *(const float4*)(krow + 32 + 8 * g);
      *((float4*)&kb[sub * 2 + 1] + 1) = *(const float4*)(krow + 48 + 8 * g);
    }
    v8f s[2] = {};
#pragma unroll
    for (int sub = 0; sub < 2; sub++) {
      s[sub] = __builtin_amdgcn_wmma_f32_16x16x32_f16(
          false, qa0, false, kb[sub * 2], (short)0, s[sub], false, false);
      s[sub] = __builtin_amdgcn_wmma_f32_16x16x32_f16(
          false, qa1, false, kb[sub * 2 + 1], (short)0, s[sub], false, false);
    }

#pragma unroll
    for (int i = 0; i < 8; i++) {
      float v0 = s[0][i] * 0.03125f;
      float v1 = s[1][i] * 0.03125f;
      if (lastb) {
        const int q = qBase + 8 * g + i;
        if (jBase + n16 > q)      v0 = -1e30f;
        if (jBase + 16 + n16 > q) v1 = -1e30f;
      }
      float mi = fmaxf(v0, v1);
#pragma unroll
      for (int msk = 1; msk < 16; msk <<= 1)
        mi = fmaxf(mi, __shfl_xor(mi, msk, 32));
      const float mnew = fmaxf(mrow[i], mi);
      const float c    = __expf(mrow[i] - mnew);
      const float p0   = __expf(v0 - mnew);
      const float p1   = __expf(v1 - mnew);
      float rs = p0 + p1;
#pragma unroll
      for (int msk = 1; msk < 16; msk <<= 1)
        rs += __shfl_xor(rs, msk, 32);
      lsum[i] = lsum[i] * c + rs;
      mrow[i] = mnew;
#pragma unroll
      for (int ct = 0; ct < 4; ct++) o[ct][i] *= c;
      plds[wave][8 * g + i][n16]      = (_Float16)p0;
      plds[wave][8 * g + i][16 + n16] = (_Float16)p1;
    }

    __builtin_amdgcn_wave_barrier();
    v16h pa;
    *(float4*)&pa       = *(const float4*)&plds[wave][n16][8 * g];
    *((float4*)&pa + 1) = *(const float4*)&plds[wave][n16][16 + 8 * g];

    v16h vb[4];
#pragma unroll
    for (int ct = 0; ct < 4; ct++) {
      const _Float16* vrow = Vt + ((size_t)b * Hh + 16 * ct + n16) * Tt + jBase;
      *(float4*)&vb[ct]       = *(const float4*)(vrow + 8 * g);
      *((float4*)&vb[ct] + 1) = *(const float4*)(vrow + 16 + 8 * g);
    }
#pragma unroll
    for (int ct = 0; ct < 4; ct++) {
      o[ct] = __builtin_amdgcn_wmma_f32_16x16x32_f16(
          false, pa, false, vb[ct], (short)0, o[ct], false, false);
    }
    __builtin_amdgcn_wave_barrier();
  }

#pragma unroll
  for (int i = 0; i < 8; i++) {
    const float inv = 1.0f / lsum[i];
    const int   t   = qBase + 8 * g + i;
    float* orow = out + ((size_t)b * Tt + t) * Hh;
#pragma unroll
    for (int ct = 0; ct < 4; ct++) orow[16 * ct + n16] = o[ct][i] * inv;
  }
}

extern "C" void kernel_launch(void* const* d_in, const int* in_sizes, int n_in,
                              void* d_out, int out_size, void* d_ws,
                              size_t ws_size, hipStream_t stream) {
  const float* x  = (const float*)d_in[0];
  const float* wk = (const float*)d_in[1];
  const float* wq = (const float*)d_in[2];
  const float* wv = (const float*)d_in[3];
  float* out = (float*)d_out;

  char* ws = (char*)d_ws;
  signed char* w8    = (signed char*)(ws + OFF_W8);
  float*       wmean = (float*)(ws + OFF_WMEAN);
  float*       amax  = (float*)(ws + OFF_AMAX);
  signed char* xq    = (signed char*)(ws + OFF_XQ);
  _Float16*    Qf    = (_Float16*)(ws + OFF_QF);
  _Float16*    Kf    = (_Float16*)(ws + OFF_KF);
  _Float16*    Vt    = (_Float16*)(ws + OFF_VT);

  wquant_kernel<<<3, 256, 0, stream>>>(wk, wq, wv, w8, wmean);
  actquant_kernel<<<BT, 256, 0, stream>>>(x, xq, amax);
  qkv_kernel<<<BT / 128, 256, WBYTES, stream>>>(xq, amax, w8, wmean, Kf, Qf, Vt);
  attn_kernel<<<Bb * (Tt / 128), 256, 0, stream>>>(Qf, Kf, Vt, out);
}
